// QuantumAttention_67508295958989
// MI455X (gfx1250) — matrix operand order repaired and checked by reading, not compiled, not run on hardware
//
#include <hip/hip_runtime.h>

// ============================================================================
// QuantumAttention for MI455X (gfx1250, wave32, WMMA)
// Pass 0: fp32 -> bf16 conversion of x + 8 weight matrices (one-shot).
// Pass 1: 6x bf16 GEMM (qkv projections), pure b128-load + wmma inner loop.
// Pass 2: LDS-tiled transpose of V into per-head [DH][SEQ] layout.
// Pass 3: complex flash attention, 24 wmma / 32 keys per query-tile wave.
// Pass 4: complex o-projection (32x32 wave tile, 16 wmma : 8 loads per chunk)
//         -> interleaved complex64 output.
// ============================================================================

typedef __bf16 bf16;
typedef __attribute__((ext_vector_type(16))) __bf16 bf16x16;
typedef __attribute__((ext_vector_type(8)))  __bf16 bf16x8;
typedef __attribute__((ext_vector_type(8)))  float  floatx8;

#define BB      4
#define SEQ     1024
#define DMODEL  1024
#define NHEAD   16
#define DHEAD   64
#define MROWS   (BB * SEQ)          // 4096
#define ATT_SCALE 0.125f            // 1/sqrt(DH)

static __device__ __forceinline__ floatx8 wmma_bf16(bf16x16 a, bf16x16 b, floatx8 c) {
  return __builtin_amdgcn_wmma_f32_16x16x32_bf16(
      /*neg_a=*/false, a, /*neg_b=*/false, b,
      /*c_mod=*/(short)0, c, /*reuse_a=*/false, /*reuse_b=*/false);
}

static __device__ __forceinline__ bf16x16 concat8(bf16x8 lo, bf16x8 hi) {
  return __builtin_shufflevector(lo, hi, 0, 1, 2, 3, 4, 5, 6, 7,
                                         8, 9, 10, 11, 12, 13, 14, 15);
}

// ---------------------------------------------------------------------------
// Fragment loaders (bf16 sources, two 16B loads each).
// A 16x32 (MxK): lane = kh*16+m; VGPR v holds k = (v>>2)*16 + kh*8 + (v&3)*2.
//   -> per lane: contiguous runs [kh*8, kh*8+8) and [16+kh*8, 16+kh*8+8).
// B 32x16 (KxN) with B[k][n] = S[n][k]: lane = kh*16+n; same k order as A:
//   per lane runs [kh*8, kh*8+8) and [16+kh*8, 16+kh*8+8) of row n.
// C/D 16x16 f32: lane -> col n = lane&15, VGPR r -> row m = r + 8*(lane>>4).
// ---------------------------------------------------------------------------

static __device__ __forceinline__ bf16x16 load_a_bf16(const bf16* __restrict__ src, int ld) {
  const int lane = (int)(threadIdx.x & 31);
  const int m    = lane & 15;
  const int kh   = lane >> 4;
  const bf16* p  = src + (long)m * ld + (kh << 3);
  const bf16x8 lo = *reinterpret_cast<const bf16x8*>(p);
  const bf16x8 hi = *reinterpret_cast<const bf16x8*>(p + 16);
  return concat8(lo, hi);
}

static __device__ __forceinline__ bf16x16 load_bT_bf16(const bf16* __restrict__ src, int ld) {
  const int lane = (int)(threadIdx.x & 31);
  const int n    = lane & 15;
  const int kh   = lane >> 4;
  const bf16* p  = src + (long)n * ld + (kh << 3);
  const bf16x8 lo = *reinterpret_cast<const bf16x8*>(p);
  const bf16x8 hi = *reinterpret_cast<const bf16x8*>(p + 16);
  return concat8(lo, hi);
}

// ---------------------------------------------------------------------------
// Pass 0: fp32 -> bf16, 8 elements per thread.
// ---------------------------------------------------------------------------
__global__ __launch_bounds__(256) void cvt_f32_bf16(
    const float* __restrict__ in, bf16* __restrict__ out, int n8) {
  const int i = (int)(blockIdx.x * 256 + threadIdx.x);
  if (i >= n8) return;
  const float4 f0 = reinterpret_cast<const float4*>(in)[2 * i];
  const float4 f1 = reinterpret_cast<const float4*>(in)[2 * i + 1];
  bf16x8 o;
  o[0] = (bf16)f0.x; o[1] = (bf16)f0.y; o[2] = (bf16)f0.z; o[3] = (bf16)f0.w;
  o[4] = (bf16)f1.x; o[5] = (bf16)f1.y; o[6] = (bf16)f1.z; o[7] = (bf16)f1.w;
  reinterpret_cast<bf16x8*>(out)[i] = o;
}

// ---------------------------------------------------------------------------
// Pass 1: Y[M,N] = X[M,K] * W[N,K]^T + bias[N]   (all bf16, f32 accum)
// Wave tile 32x64 (8 wmma per k-chunk); block 256 -> 128x128; grid (32, 8).
// Plain loop body: the scheduler interleaves the 12 b128 loads with the 8
// WMMAs on its own (manual double-buffering regressed into v_mov/v_nop).
// ---------------------------------------------------------------------------
__global__ __launch_bounds__(256) void gemm_bf16_bias(
    const bf16* __restrict__ X, const bf16* __restrict__ W,
    const float* __restrict__ bias, bf16* __restrict__ Y) {
  const int wave = (int)(threadIdx.x >> 5);
  const int lane = (int)(threadIdx.x & 31);
  const int m0 = blockIdx.x * 128 + (wave & 3) * 32;
  const int n0 = blockIdx.y * 128 + (wave >> 2) * 64;

  floatx8 c[2][4] = {};
  for (int k = 0; k < DMODEL; k += 32) {
    const bf16x16 a0 = load_a_bf16(X + (long)m0 * DMODEL + k, DMODEL);
    const bf16x16 a1 = load_a_bf16(X + (long)(m0 + 16) * DMODEL + k, DMODEL);
    bf16x16 b[4];
#pragma unroll
    for (int j = 0; j < 4; ++j)
      b[j] = load_bT_bf16(W + (long)(n0 + 16 * j) * DMODEL + k, DMODEL);
#pragma unroll
    for (int j = 0; j < 4; ++j) {
      c[0][j] = wmma_bf16(a0, b[j], c[0][j]);
      c[1][j] = wmma_bf16(a1, b[j], c[1][j]);
    }
  }

  const int nloc  = lane & 15;
  const int rbase = (lane >> 4) << 3;
#pragma unroll
  for (int j = 0; j < 4; ++j) {
    const float bn = bias[n0 + 16 * j + nloc];
#pragma unroll
    for (int i = 0; i < 2; ++i) {
#pragma unroll
      for (int r = 0; r < 8; ++r) {
        Y[(long)(m0 + 16 * i + rbase + r) * DMODEL + n0 + 16 * j + nloc] =
            (bf16)(c[i][j][r] + bn);
      }
    }
  }
}

// ---------------------------------------------------------------------------
// Pass 2: per-head transpose V[b*SEQ + j][h*DH + d] -> Vt[(bh*DH + d)*SEQ + j]
// 64x64 tile per block through LDS; fully coalesced both directions.
// ---------------------------------------------------------------------------
__global__ __launch_bounds__(256) void transpose_v(
    const bf16* __restrict__ V, bf16* __restrict__ Vt) {
  __shared__ bf16 t[64][65];
  const int j0 = blockIdx.x * 64;
  const int bh = blockIdx.y;
  const int b  = bh / NHEAD;
  const int h  = bh - b * NHEAD;
  const long rowBase = (long)b * SEQ;
  const int  colBase = h * DHEAD;

  const int dr = (int)(threadIdx.x & 63);
  const int jr = (int)(threadIdx.x >> 6);
#pragma unroll
  for (int r = 0; r < 16; ++r) {
    const int j = jr * 16 + r;
    t[j][dr] = V[(rowBase + j0 + j) * DMODEL + colBase + dr];
  }
  __syncthreads();
  const int jw = (int)(threadIdx.x & 63);
  const int dw = (int)(threadIdx.x >> 6);
#pragma unroll
  for (int r = 0; r < 16; ++r) {
    const int d = dw * 16 + r;
    Vt[((long)bh * DHEAD + d) * SEQ + j0 + jw] = t[jw][d];
  }
}

// ---------------------------------------------------------------------------
// Pass 3: complex flash attention, one wave per (16-query tile, b, h).
//   sim_re = qr*kr^T + qi*ki^T ; sim_im = qi*kr^T - qr*ki^T
//   attn = softmax(SCALE * |sim|) (online, 32-key chunks), O = attn @ (vr,vi).
// q/k are bf16 [4096,1024]; v is pre-transposed bf16 [64][64][1024].
// ---------------------------------------------------------------------------
__global__ __launch_bounds__(32) void qattn(
    const bf16* __restrict__ Qr, const bf16* __restrict__ Qi,
    const bf16* __restrict__ Kr, const bf16* __restrict__ Ki,
    const bf16* __restrict__ Vtr, const bf16* __restrict__ Vti,
    bf16* __restrict__ Or, bf16* __restrict__ Oi) {
  __shared__ __align__(16) bf16 ldsP[16][32];

  const int lane = (int)(threadIdx.x & 31);
  const int qt   = blockIdx.x * 16;
  const int bh   = blockIdx.y;
  const int b    = bh / NHEAD;
  const int h    = bh - b * NHEAD;
  const long rowBase = (long)b * SEQ;
  const int  colBase = h * DHEAD;
  const bf16* vtr = Vtr + (long)bh * DHEAD * SEQ;
  const bf16* vti = Vti + (long)bh * DHEAD * SEQ;

  bf16x16 aqr[2], aqi[2];
#pragma unroll
  for (int c = 0; c < 2; ++c) {
    aqr[c] = load_a_bf16(Qr + (rowBase + qt) * DMODEL + colBase + c * 32, DMODEL);
    aqi[c] = load_a_bf16(Qi + (rowBase + qt) * DMODEL + colBase + c * 32, DMODEL);
  }

  floatx8 ore0 = {}, ore1 = {}, ore2 = {}, ore3 = {};
  floatx8 oim0 = {}, oim1 = {}, oim2 = {}, oim3 = {};
  float mrow[8], lrow[8];
#pragma unroll
  for (int r = 0; r < 8; ++r) { mrow[r] = -3.0e38f; lrow[r] = 0.0f; }

  for (int j0 = 0; j0 < SEQ; j0 += 32) {
    floatx8 mag[2];
#pragma unroll
    for (int jt = 0; jt < 2; ++jt) {
      floatx8 sre = {}, s1 = {}, s2 = {};
      const bf16* kbr = Kr + (rowBase + j0 + jt * 16) * DMODEL + colBase;
      const bf16* kbi = Ki + (rowBase + j0 + jt * 16) * DMODEL + colBase;
#pragma unroll
      for (int c = 0; c < 2; ++c) {
        const bf16x16 bkr = load_bT_bf16(kbr + c * 32, DMODEL);
        const bf16x16 bki = load_bT_bf16(kbi + c * 32, DMODEL);
        sre = wmma_bf16(aqr[c], bkr, sre);   // qr*kr
        sre = wmma_bf16(aqi[c], bki, sre);   // + qi*ki
        s1  = wmma_bf16(aqi[c], bkr, s1);    // qi*kr
        s2  = wmma_bf16(aqr[c], bki, s2);    // qr*ki   (im = s1 - s2)
      }
#pragma unroll
      for (int r = 0; r < 8; ++r) {
        const float re = sre[r];
        const float im = s1[r] - s2[r];
        mag[jt][r] = ATT_SCALE * __fsqrt_rn(re * re + im * im);
      }
    }

    // Online softmax over 32 fresh columns; row r lives in a 16-lane group.
#pragma unroll
    for (int r = 0; r < 8; ++r) {
      float rmax = fmaxf(mag[0][r], mag[1][r]);
#pragma unroll
      for (int off = 8; off >= 1; off >>= 1)
        rmax = fmaxf(rmax, __shfl_xor(rmax, off));
      const float mnew  = fmaxf(mrow[r], rmax);
      const float alpha = __expf(mrow[r] - mnew);
      mrow[r] = mnew;

      const float p0 = __expf(mag[0][r] - mnew);
      const float p1 = __expf(mag[1][r] - mnew);
      float psum = p0 + p1;
#pragma unroll
      for (int off = 8; off >= 1; off >>= 1)
        psum += __shfl_xor(psum, off);
      lrow[r] = lrow[r] * alpha + psum;

      ore0[r] *= alpha; ore1[r] *= alpha; ore2[r] *= alpha; ore3[r] *= alpha;
      oim0[r] *= alpha; oim1[r] *= alpha; oim2[r] *= alpha; oim3[r] *= alpha;

      const int m = r + ((lane >> 4) << 3);
      const int n = lane & 15;
      ldsP[m][n]      = (bf16)p0;
      ldsP[m][n + 16] = (bf16)p1;
    }

    // Re-read P as an A fragment (two 16B DS loads; DS in-order per wave).
    bf16x16 pa;
    {
      const int m  = lane & 15;
      const int kh = lane >> 4;
      const bf16* lp = &ldsP[m][kh << 3];
      const bf16x8 lo = *reinterpret_cast<const bf16x8*>(lp);
      const bf16x8 hi = *reinterpret_cast<const bf16x8*>(lp + 16);
      pa = concat8(lo, hi);
    }

    // O += P @ V via transposed V: B[k=j][n=d] = Vt[d][j0+k] -> contiguous.
    __builtin_prefetch(vtr + j0 + 32, 0, 3);
    ore0 = wmma_bf16(pa, load_bT_bf16(vtr + (long)0  * SEQ + j0, SEQ), ore0);
    ore1 = wmma_bf16(pa, load_bT_bf16(vtr + (long)16 * SEQ + j0, SEQ), ore1);
    ore2 = wmma_bf16(pa, load_bT_bf16(vtr + (long)32 * SEQ + j0, SEQ), ore2);
    ore3 = wmma_bf16(pa, load_bT_bf16(vtr + (long)48 * SEQ + j0, SEQ), ore3);
    oim0 = wmma_bf16(pa, load_bT_bf16(vti + (long)0  * SEQ + j0, SEQ), oim0);
    oim1 = wmma_bf16(pa, load_bT_bf16(vti + (long)16 * SEQ + j0, SEQ), oim1);
    oim2 = wmma_bf16(pa, load_bT_bf16(vti + (long)32 * SEQ + j0, SEQ), oim2);
    oim3 = wmma_bf16(pa, load_bT_bf16(vti + (long)48 * SEQ + j0, SEQ), oim3);
  }

  const int n  = lane & 15;
  const int mb = qt + ((lane >> 4) << 3);
#pragma unroll
  for (int r = 0; r < 8; ++r) {
    const float inv = 1.0f / lrow[r];
    const long row = (rowBase + mb + r) * DMODEL + colBase;
    Or[row + 0  + n] = (bf16)(ore0[r] * inv);
    Or[row + 16 + n] = (bf16)(ore1[r] * inv);
    Or[row + 32 + n] = (bf16)(ore2[r] * inv);
    Or[row + 48 + n] = (bf16)(ore3[r] * inv);
    Oi[row + 0  + n] = (bf16)(oim0[r] * inv);
    Oi[row + 16 + n] = (bf16)(oim1[r] * inv);
    Oi[row + 32 + n] = (bf16)(oim2[r] * inv);
    Oi[row + 48 + n] = (bf16)(oim3[r] * inv);
  }
}

// ---------------------------------------------------------------------------
// Pass 4: complex output projection to interleaved complex64.
//   re = ar*Wr^T - ai*Wi^T + br ;  im = ar*Wi^T + ai*Wr^T + bi
// Wave tile 32x32: 8 fragment loads feed 16 WMMAs per k-chunk (complex reuse).
// Block 256 -> 128x64 tile; grid (64, 32).
// ---------------------------------------------------------------------------
__global__ __launch_bounds__(256) void oproj(
    const bf16* __restrict__ Ar, const bf16* __restrict__ Ai,
    const bf16* __restrict__ Wr, const bf16* __restrict__ Wi,
    const float* __restrict__ br, const float* __restrict__ bi,
    float2* __restrict__ out) {
  const int wave = (int)(threadIdx.x >> 5);
  const int lane = (int)(threadIdx.x & 31);
  const int m0 = blockIdx.x * 128 + (wave & 3) * 32;
  const int n0 = blockIdx.y * 64  + (wave >> 2) * 32;

  floatx8 c1[2][2] = {}, c2[2][2] = {}, cim[2][2] = {};
  for (int k = 0; k < DMODEL; k += 32) {
    bf16x16 ar[2], ai[2], wr[2], wi[2];
#pragma unroll
    for (int i = 0; i < 2; ++i) {
      ar[i] = load_a_bf16(Ar + (long)(m0 + 16 * i) * DMODEL + k, DMODEL);
      ai[i] = load_a_bf16(Ai + (long)(m0 + 16 * i) * DMODEL + k, DMODEL);
      wr[i] = load_bT_bf16(Wr + (long)(n0 + 16 * i) * DMODEL + k, DMODEL);
      wi[i] = load_bT_bf16(Wi + (long)(n0 + 16 * i) * DMODEL + k, DMODEL);
    }
#pragma unroll
    for (int i = 0; i < 2; ++i) {
#pragma unroll
      for (int j = 0; j < 2; ++j) {
        c1[i][j]  = wmma_bf16(ar[i], wr[j], c1[i][j]);   // ar*Wr
        c2[i][j]  = wmma_bf16(ai[i], wi[j], c2[i][j]);   // ai*Wi (re = c1-c2)
        cim[i][j] = wmma_bf16(ar[i], wi[j], cim[i][j]);  // ar*Wi
        cim[i][j] = wmma_bf16(ai[i], wr[j], cim[i][j]);  // + ai*Wr
      }
    }
  }

  const int nloc = lane & 15;
  const int rbase = (lane >> 4) << 3;
#pragma unroll
  for (int j = 0; j < 2; ++j) {
    const int n = n0 + 16 * j + nloc;
    const float vbr = br[n];
    const float vbi = bi[n];
#pragma unroll
    for (int i = 0; i < 2; ++i) {
#pragma unroll
      for (int r = 0; r < 8; ++r) {
        float2 o;
        o.x = c1[i][j][r] - c2[i][j][r] + vbr;
        o.y = cim[i][j][r] + vbi;
        out[(long)(m0 + 16 * i + rbase + r) * DMODEL + n] = o;
      }
    }
  }
}

// ---------------------------------------------------------------------------
// Host-side launch. Workspace (bf16 elements):
//   xb 4M | 8 weight planes 1M each | q/k re+im 4x4M | v re+im 2x4M
//   vt re+im 2x4M | ao re+im 2x4M   -> 52M elems = 104 MB
// ---------------------------------------------------------------------------
extern "C" void kernel_launch(void* const* d_in, const int* in_sizes, int n_in,
                              void* d_out, int out_size, void* d_ws, size_t ws_size,
                              hipStream_t stream) {
  (void)in_sizes; (void)n_in; (void)out_size; (void)ws_size;

  const float* x   = (const float*)d_in[0];
  const float* qWr = (const float*)d_in[1];
  const float* qWi = (const float*)d_in[2];
  const float* qbr = (const float*)d_in[3];
  const float* qbi = (const float*)d_in[4];
  const float* kWr = (const float*)d_in[5];
  const float* kWi = (const float*)d_in[6];
  const float* kbr = (const float*)d_in[7];
  const float* kbi = (const float*)d_in[8];
  const float* vWr = (const float*)d_in[9];
  const float* vWi = (const float*)d_in[10];
  const float* vbr = (const float*)d_in[11];
  const float* vbi = (const float*)d_in[12];
  const float* oWr = (const float*)d_in[13];
  const float* oWi = (const float*)d_in[14];
  const float* obr = (const float*)d_in[15];
  const float* obi = (const float*)d_in[16];

  bf16* ws = (bf16*)d_ws;
  const long XPLANE = (long)MROWS * DMODEL;     // 4M
  const long WPLANE = (long)DMODEL * DMODEL;    // 1M
  bf16* xb    = ws;                 // 4M
  bf16* wqrb  = xb + XPLANE;        // 8 x 1M weight planes
  bf16* wqib  = wqrb + WPLANE;
  bf16* wkrb  = wqib + WPLANE;
  bf16* wkib  = wkrb + WPLANE;
  bf16* wvrb  = wkib + WPLANE;
  bf16* wvib  = wvrb + WPLANE;
  bf16* worb  = wvib + WPLANE;
  bf16* woib  = worb + WPLANE;
  bf16* qre   = woib + WPLANE;      // 10 x 4M activation planes
  bf16* qim   = qre + XPLANE;
  bf16* kre   = qim + XPLANE;
  bf16* kim   = kre + XPLANE;
  bf16* vre   = kim + XPLANE;
  bf16* vim   = vre + XPLANE;
  bf16* vtr   = vim + XPLANE;
  bf16* vti   = vtr + XPLANE;
  bf16* aor   = vti + XPLANE;
  bf16* aoi   = aor + XPLANE;

  // --- Pass 0: convert fp32 operands to bf16 once ---
  const int X8 = (int)(XPLANE / 8), W8 = (int)(WPLANE / 8);
  cvt_f32_bf16<<<(X8 + 255) / 256, 256, 0, stream>>>(x,   xb,   X8);
  cvt_f32_bf16<<<(W8 + 255) / 256, 256, 0, stream>>>(qWr, wqrb, W8);
  cvt_f32_bf16<<<(W8 + 255) / 256, 256, 0, stream>>>(qWi, wqib, W8);
  cvt_f32_bf16<<<(W8 + 255) / 256, 256, 0, stream>>>(kWr, wkrb, W8);
  cvt_f32_bf16<<<(W8 + 255) / 256, 256, 0, stream>>>(kWi, wkib, W8);
  cvt_f32_bf16<<<(W8 + 255) / 256, 256, 0, stream>>>(vWr, wvrb, W8);
  cvt_f32_bf16<<<(W8 + 255) / 256, 256, 0, stream>>>(vWi, wvib, W8);
  cvt_f32_bf16<<<(W8 + 255) / 256, 256, 0, stream>>>(oWr, worb, W8);
  cvt_f32_bf16<<<(W8 + 255) / 256, 256, 0, stream>>>(oWi, woib, W8);

  // --- Pass 1: QKV projections (6 bf16 GEMMs) ---
  const dim3 gGemm(MROWS / 128, DMODEL / 128, 1);
  gemm_bf16_bias<<<gGemm, 256, 0, stream>>>(xb, wqrb, qbr, qre);
  gemm_bf16_bias<<<gGemm, 256, 0, stream>>>(xb, wqib, qbi, qim);
  gemm_bf16_bias<<<gGemm, 256, 0, stream>>>(xb, wkrb, kbr, kre);
  gemm_bf16_bias<<<gGemm, 256, 0, stream>>>(xb, wkib, kbi, kim);
  gemm_bf16_bias<<<gGemm, 256, 0, stream>>>(xb, wvrb, vbr, vre);
  gemm_bf16_bias<<<gGemm, 256, 0, stream>>>(xb, wvib, vbi, vim);

  // --- Pass 2: transpose V (re, im) into per-head [DH][SEQ] ---
  const dim3 gTr(SEQ / 64, BB * NHEAD, 1);
  transpose_v<<<gTr, 256, 0, stream>>>(vre, vtr);
  transpose_v<<<gTr, 256, 0, stream>>>(vim, vti);

  // --- Pass 3: complex flash attention ---
  qattn<<<dim3(SEQ / 16, BB * NHEAD, 1), 32, 0, stream>>>(
      qre, qim, kre, kim, vtr, vti, aor, aoi);

  // --- Pass 4: output projection to interleaved complex64 ---
  oproj<<<dim3(MROWS / 64, DMODEL / 32, 1), 256, 0, stream>>>(
      aor, aoi, worb, woib, obr, obi, (float2*)d_out);
}
